// SimpleRNN_1769526526608
// MI455X (gfx1250) — matrix operand order repaired and checked by reading, not compiled, not run on hardware
//
#include <hip/hip_runtime.h>
#include <math.h>

// ---------------- types for WMMA fragments ----------------
typedef __attribute__((ext_vector_type(16))) __bf16        v16bf;
typedef __attribute__((ext_vector_type(8)))  float         v8f;
typedef __attribute__((ext_vector_type(4)))  unsigned int  u32x4;
typedef __attribute__((ext_vector_type(4)))  float         f32x4;

union ABfrag { u32x4 q[2]; v16bf bf; };   // 32 bytes: 16 bf16 per lane

// round-to-nearest-even f32 -> bf16
static __device__ __forceinline__ unsigned short f2bf(float f) {
    unsigned int u = __builtin_bit_cast(unsigned int, f);
    u += 0x7FFFu + ((u >> 16) & 1u);
    return (unsigned short)(u >> 16);
}

static __device__ __forceinline__ float tanh_fast(float x) {
    float ax = fabsf(x);
    float e  = __expf(-2.0f * ax);          // v_exp_f32 path
    float t  = (1.0f - e) / (1.0f + e);
    return copysignf(t, x);
}

#define WMMA_BF16(A, B, C) __builtin_amdgcn_wmma_f32_16x16x32_bf16( \
        false, (A).bf, false, (B).bf, (short)0, (C), false, false)

// ---------------- elementwise f32 -> bf16 (vector of 8) ----------------
__global__ void __launch_bounds__(256) k_cvt_bf16(const float* __restrict__ in,
                                                  unsigned short* __restrict__ out,
                                                  size_t n8) {  // n8 = n/8
    size_t i = (size_t)blockIdx.x * blockDim.x + threadIdx.x;
    if (i >= n8) return;
    size_t base = i * 8;
    f32x4 a = ((const f32x4*)(in + base))[0];
    f32x4 b = ((const f32x4*)(in + base))[1];
    u32x4 o;
    o.x = (unsigned)f2bf(a.x) | ((unsigned)f2bf(a.y) << 16);
    o.y = (unsigned)f2bf(a.z) | ((unsigned)f2bf(a.w) << 16);
    o.z = (unsigned)f2bf(b.x) | ((unsigned)f2bf(b.y) << 16);
    o.w = (unsigned)f2bf(b.z) | ((unsigned)f2bf(b.w) << 16);
    ((u32x4*)(out + base))[0] = o;
}

// ---------------- fused bias: bsum = b_xh + b_hh ----------------
__global__ void __launch_bounds__(256) k_bias_sum(const float* __restrict__ a,
                                                  const float* __restrict__ b,
                                                  float* __restrict__ o, int n) {
    int i = blockIdx.x * blockDim.x + threadIdx.x;
    if (i < n) o[i] = a[i] + b[i];
}

// ---------------- GEMM: out[M,512] = A[M,512](bf16) @ W[512,512](bf16, row=n)^T + bias ----------------
// per-wave tile: 32 (M) x 64 (N); K=512 in 16 WMMA steps; software-pipelined fragment loads.
__global__ void __launch_bounds__(256) k_gemm_bf16(const unsigned short* __restrict__ A,
                                                   const unsigned short* __restrict__ W,
                                                   const float* __restrict__ bias,
                                                   float* __restrict__ out,
                                                   int M) {
    const int K = 512, N = 512;
    int wid  = (blockIdx.x * blockDim.x + threadIdx.x) >> 5;
    int lane = threadIdx.x & 31;
    int mt = wid >> 3;        // 32-row M tile
    int ng = wid & 7;         // 64-col N group
    if (mt * 32 >= M) return; // wave-uniform: EXEC stays all-ones for WMMA
    int l15 = lane & 15;
    int hi  = lane >> 4;

    const unsigned short* Abase0 = A + (size_t)(mt * 32 + l15) * K + hi * 16;
    const unsigned short* Abase1 = Abase0 + (size_t)16 * K;
    const unsigned short* Wbase  = W + (size_t)(ng * 64 + l15) * K + hi * 16;

    auto loadA = [&](int kb, ABfrag af[2]) {
        const u32x4* p0 = (const u32x4*)(Abase0 + kb * 32);
        af[0].q[0] = p0[0]; af[0].q[1] = p0[1];
        const u32x4* p1 = (const u32x4*)(Abase1 + kb * 32);
        af[1].q[0] = p1[0]; af[1].q[1] = p1[1];
    };
    auto loadB = [&](int kb, ABfrag bf[4]) {
#pragma unroll
        for (int j = 0; j < 4; ++j) {
            const u32x4* p = (const u32x4*)(Wbase + (size_t)j * 16 * K + kb * 32);
            bf[j].q[0] = p[0]; bf[j].q[1] = p[1];
        }
    };

    v8f acc[8] = {};                       // [m-half][j] -> 8 independent chains
    ABfrag a[2][2], b[2][4];
    loadA(0, a[0]); loadB(0, b[0]);
#pragma unroll
    for (int kb = 0; kb < 16; ++kb) {
        int cur = kb & 1, nxt = cur ^ 1;
        if (kb < 15) { loadA(kb + 1, a[nxt]); loadB(kb + 1, b[nxt]); }
#pragma unroll
        for (int j = 0; j < 4; ++j) {
            acc[j]     = WMMA_BF16(a[cur][0], b[cur][j], acc[j]);
            acc[4 + j] = WMMA_BF16(a[cur][1], b[cur][j], acc[4 + j]);
        }
    }
#pragma unroll
    for (int j = 0; j < 4; ++j) {
        int n = ng * 64 + j * 16 + l15;
        float bv = bias[n];
#pragma unroll
        for (int mh = 0; mh < 2; ++mh)
#pragma unroll
            for (int i = 0; i < 8; ++i) {
                int m = mt * 32 + mh * 16 + hi * 8 + i;   // D: lanes 16-31 = rows 8..15
                out[(size_t)m * N + n] = acc[mh * 4 + j][i] + bv;
            }
    }
}

// ---------------- sequential scan, batch-partitioned: 4 WGs x 16 batch rows ----------------
// h double-buffered in LDS (bf16); W_hh streamed from L2; pipelined loads; 8 acc chains.
__global__ void __launch_bounds__(256, 1) k_rnn_scan(const unsigned short* __restrict__ Whh,  // 512x512 bf16 (n,k)
                                                     const float* __restrict__ xp,            // (B,T,512) f32
                                                     const float* __restrict__ h0,            // (B,512) f32
                                                     unsigned short* __restrict__ hs,         // (B,T,512) bf16
                                                     float* __restrict__ hn) {                // (B,512) f32
    const int T = 1024, H = 512;
    __shared__ __align__(16) unsigned short hbuf[2][16 * 512];   // 32 KB

    int b0 = blockIdx.x * 16;
    for (int idx = threadIdx.x; idx < 16 * 512; idx += 256) {
        int r = idx >> 9, c = idx & 511;
        hbuf[0][idx] = f2bf(h0[(size_t)(b0 + r) * H + c]);
    }
    __syncthreads();

    int w = threadIdx.x >> 5, lane = threadIdx.x & 31;
    int l15 = lane & 15, hi = lane >> 4;

    const unsigned short* Wbase = Whh + (size_t)(w * 64 + l15) * H + hi * 16;

    for (int t = 0; t < T; ++t) {
        const unsigned short* hsrc = hbuf[t & 1];
        unsigned short*       hdst = hbuf[(t + 1) & 1];

        // hoisted xp loads for this step (consumed after the WMMA pipeline)
        float xpv[32];
#pragma unroll
        for (int j = 0; j < 4; ++j)
#pragma unroll
            for (int i = 0; i < 8; ++i) {
                int m = hi * 8 + i;
                size_t row = (size_t)(b0 + m) * T + t;
                xpv[j * 8 + i] = xp[row * H + w * 64 + j * 16 + l15];
            }

        // one wave-wide prefetch of next step's xp tile (lane -> 128B chunk)
        if (t < T - 1) {
            size_t prow = (size_t)(b0 + l15) * T + (t + 1);
            const char* pp = (const char*)(xp + prow * H + w * 64) + hi * 128;
            __builtin_prefetch(pp, 0, 3);
        }

        auto loadA = [&](int kb, ABfrag& af) {
            const u32x4* p = (const u32x4*)(hsrc + l15 * 512 + kb * 32 + hi * 16);
            af.q[0] = p[0]; af.q[1] = p[1];
        };
        auto loadB = [&](int kb, ABfrag bf[4]) {
#pragma unroll
            for (int j = 0; j < 4; ++j) {
                const u32x4* p = (const u32x4*)(Wbase + (size_t)j * 16 * H + kb * 32);
                bf[j].q[0] = p[0]; bf[j].q[1] = p[1];
            }
        };

        v8f acc[8] = {};                   // [kb-parity][j] -> 8 chains of depth 8
        ABfrag a[2], b[2][4];
        loadA(0, a[0]); loadB(0, b[0]);
#pragma unroll
        for (int kb = 0; kb < 16; ++kb) {
            int cur = kb & 1, nxt = cur ^ 1;
            if (kb < 15) { loadA(kb + 1, a[nxt]); loadB(kb + 1, b[nxt]); }
#pragma unroll
            for (int j = 0; j < 4; ++j)
                acc[j + cur * 4] = WMMA_BF16(a[cur], b[cur][j], acc[j + cur * 4]);
        }

        // combine chains, add xp, tanh, write next-h (LDS) + hs (global bf16) + final h_n
#pragma unroll
        for (int j = 0; j < 4; ++j) {
            v8f s = acc[j] + acc[j + 4];
            int n = w * 64 + j * 16 + l15;
#pragma unroll
            for (int i = 0; i < 8; ++i) {
                int m = hi * 8 + i;
                int b = b0 + m;
                size_t row = (size_t)b * T + t;
                float h = tanh_fast(s[i] + xpv[j * 8 + i]);
                unsigned short hb = f2bf(h);
                hdst[m * 512 + n] = hb;
                hs[row * H + n] = hb;
                if (t == T - 1) hn[(size_t)b * H + n] = h;
            }
        }
        __syncthreads();
    }
}

// ---------------- launch ----------------
extern "C" void kernel_launch(void* const* d_in, const int* in_sizes, int n_in,
                              void* d_out, int out_size, void* d_ws, size_t ws_size,
                              hipStream_t stream) {
    const float* x   = (const float*)d_in[0];
    const float* h0  = (const float*)d_in[1];
    const float* Wxh = (const float*)d_in[2];
    const float* bxh = (const float*)d_in[3];
    const float* Whh = (const float*)d_in[4];
    const float* bhh = (const float*)d_in[5];
    const float* Why = (const float*)d_in[6];
    const float* bhy = (const float*)d_in[7];

    const size_t B = 64, T = 1024, H = 512;
    const size_t BT = B * T;          // 65536 rows
    const size_t XE = BT * H;         // 33,554,432 elements

    char* ws = (char*)d_ws;
    unsigned short* xbf  = (unsigned short*)ws;               // XE bf16
    unsigned short* hsbf = (unsigned short*)(ws + XE * 2);    // XE bf16
    unsigned short* WxhB = (unsigned short*)(ws + XE * 4);    // 512x512 bf16
    unsigned short* WhhB = WxhB + H * H;
    unsigned short* WhyB = WhhB + H * H;
    float*          bsum = (float*)(WhyB + H * H);            // 512 f32

    float* xp = (float*)d_out;                // staged in d_out, overwritten by final GEMM
    float* hn = (float*)d_out + XE;           // h_n tail of d_out

    // 1) conversions to bf16
    k_cvt_bf16<<<(unsigned)((XE / 8 + 255) / 256), 256, 0, stream>>>(x, xbf, XE / 8);
    k_cvt_bf16<<<(unsigned)((H * H / 8 + 255) / 256), 256, 0, stream>>>(Wxh, WxhB, H * H / 8);
    k_cvt_bf16<<<(unsigned)((H * H / 8 + 255) / 256), 256, 0, stream>>>(Whh, WhhB, H * H / 8);
    k_cvt_bf16<<<(unsigned)((H * H / 8 + 255) / 256), 256, 0, stream>>>(Why, WhyB, H * H / 8);
    k_bias_sum<<<2, 256, 0, stream>>>(bxh, bhh, bsum, (int)H);

    // 2) xp = x @ Wxh^T + (b_xh + b_hh)   -> d_out (f32)
    unsigned gemmBlocks = (unsigned)(BT / 32);   // waves = (M/32)*8, 8 waves/block
    k_gemm_bf16<<<gemmBlocks, 256, 0, stream>>>(xbf, WxhB, bsum, xp, (int)BT);

    // 3) sequential scan (4 WGs, batch-partitioned, no inter-WG sync)
    k_rnn_scan<<<4, 256, 0, stream>>>(WhhB, xp, h0, hsbf, hn);

    // 4) out = hs @ Why^T + b_hy          -> d_out (overwrites xp in place)
    k_gemm_bf16<<<gemmBlocks, 256, 0, stream>>>(hsbf, WhyB, bhy, xp, (int)BT);
}
